// Sub4BitLinear_73289321939386
// MI455X (gfx1250) — matrix operand order repaired and checked by reading, not compiled, not run on hardware
//
#include <hip/hip_runtime.h>

// ---------------------------------------------------------------------------
// Sub-4bit (3-bit codebook) linear layer:  y = x @ dequant(W).T
//   x:          [8192, 4096] f32   (B*S rows)     d_in[0]
//   quant_grid: [4096, 8]    f32                  d_in[1]
//   codes:      [4096, 4096] i32 (0..7)           d_in[2]
//   y:          [8192, 4096] f32                  d_out
//
// Phase 1: dequant W once -> bf16 hi/lo split tables in d_ws (67 MB).
//          Bandwidth bound: 67MB read + 67MB write ~ 6us @ 23.3 TB/s.
// Phase 2: compute-bound GEMM (AI ~820 flop/byte) with
//          v_wmma_f32_16x16x32_bf16 and the 3-term bf16 split
//          (x_hi*w_hi + x_hi*w_lo + x_lo*w_hi, f32 accum) for ~fp32 accuracy
//          at ~2.7x the fp32-WMMA (16x16x4) rate.
//          B tiles stream via global_load_async_to_lds_b128 (double
//          buffered, s_wait_asynccnt); A tiles convert f32->hi/lo on the fly
//          (hidden under the WMMA/TRANS pipe).
// ---------------------------------------------------------------------------

typedef __attribute__((ext_vector_type(16))) __bf16 v16bf;
typedef __attribute__((ext_vector_type(8)))  float  v8f;

#define M_TOT 8192
#define N_TOT 4096
#define K_TOT 4096
#define BM 128
#define BN 128
#define BK 32
#define APAD 40   // halfs per LDS row (32 data + 8 pad) -> 80B stride, bank-spread

struct U4 { unsigned x, y, z, w; };            // trivial 16B chunk (union-safe)
struct H4 { unsigned short a, b, c, d; };      // packed 8B half store
union FragU { struct { U4 a, b; } q; v16bf v; };

__device__ __forceinline__ unsigned short bf16_rne(float v) {
  unsigned u = __builtin_bit_cast(unsigned, v);
  return (unsigned short)((u + 0x7FFFu + ((u >> 16) & 1u)) >> 16);
}

__device__ __forceinline__ void split_bf16(float v, unsigned short& hi, unsigned short& lo) {
  unsigned short h = bf16_rne(v);
  hi = h;
  float hf = __builtin_bit_cast(float, ((unsigned)h) << 16);
  lo = bf16_rne(v - hf);
}

// ---------------------------------------------------------------------------
// Phase 1: W[n,k] = codebook[n][codes[n,k]] -> whi/wlo bf16 tables.
// One block per output row n; codebook pre-split into 8 packed entries in LDS.
// ---------------------------------------------------------------------------
__global__ __launch_bounds__(256)
void dequant_split_kernel(const float* __restrict__ qg,
                          const int*   __restrict__ codes,
                          unsigned short* __restrict__ whi,
                          unsigned short* __restrict__ wlo) {
  __shared__ unsigned cb32[8];                 // hi | (lo << 16)
  const int n = blockIdx.x;
  if (threadIdx.x < 8) {
    float q = qg[(size_t)n * 8 + threadIdx.x];
    unsigned short h, l;
    split_bf16(q, h, l);
    cb32[threadIdx.x] = (unsigned)h | ((unsigned)l << 16);
  }
  __syncthreads();

  const int* crow = codes + (size_t)n * K_TOT;
  const size_t base = (size_t)n * K_TOT;
#pragma unroll
  for (int j = 0; j < 4; ++j) {
    int k = (j * 256 + threadIdx.x) * 4;       // 4096 = 256 thr * 4 iters * 4 elems
    const int4 c = *reinterpret_cast<const int4*>(crow + k);
    unsigned v0 = cb32[c.x & 7], v1 = cb32[c.y & 7];
    unsigned v2 = cb32[c.z & 7], v3 = cb32[c.w & 7];
    H4 h = { (unsigned short)v0, (unsigned short)v1,
             (unsigned short)v2, (unsigned short)v3 };
    H4 l = { (unsigned short)(v0 >> 16), (unsigned short)(v1 >> 16),
             (unsigned short)(v2 >> 16), (unsigned short)(v3 >> 16) };
    *reinterpret_cast<H4*>(whi + base + k) = h;
    *reinterpret_cast<H4*>(wlo + base + k) = l;
  }
}

// ---------------------------------------------------------------------------
// Fragment loads (CDNA5 ISA VGPR layouts)
// A (16x32 bf16, MxK): lanes 0-15 row M=lane, K {0..7,16..23};
//                      lanes 16-31 row M=lane-16, K {8..15,24..31}
// ---------------------------------------------------------------------------
__device__ __forceinline__ v16bf load_frag_A(const unsigned short* s, int rowBase, int lane) {
  int row = rowBase + (lane & 15);
  int k0  = (lane < 16) ? 0 : 8;
  const unsigned short* p = s + row * APAD;
  FragU u;
  u.q.a = *(const U4*)(p + k0);
  u.q.b = *(const U4*)(p + k0 + 16);
  return u.v;
}

// B (32x16 bf16, KxN): lanes 0-15 col N=lane K {0..7,16..23}; lanes 16-31 K {8..15,24..31}
__device__ __forceinline__ v16bf load_frag_B(const unsigned short* s, int rowBase, int lane) {
  int row = rowBase + (lane & 15);
  int k0  = (lane < 16) ? 0 : 8;
  const unsigned short* p = s + row * APAD + k0;
  FragU u;
  u.q.a = *(const U4*)(p);
  u.q.b = *(const U4*)(p + 16);
  return u.v;
}

__device__ __forceinline__ v8f wmma_bf16(v16bf a, v16bf b, v8f c) {
  return __builtin_amdgcn_wmma_f32_16x16x32_bf16(false, a, false, b, (short)0, c, false, false);
}

// Async global->LDS copy of one 128x32-half tile (2 b128 instructions/wave).
// Per-lane LDS address keeps the APAD bank-spreading layout.
__device__ __forceinline__ void async_copy_tile(const unsigned short* __restrict__ gsrc,
                                                unsigned short* ldsDst,
                                                int n0, int kt, int t) {
#pragma unroll
  for (int j = 0; j < 2; ++j) {
    int chunk = j * 256 + t;                   // 0..511 = 128 rows * 4 chunks
    int row   = chunk >> 2;
    int koff  = (chunk & 3) * 8;               // halfs; 16B per chunk
    unsigned lds_addr = (unsigned)(uintptr_t)(ldsDst + row * APAD + koff);
    unsigned long long ga =
        (unsigned long long)(uintptr_t)(gsrc + (size_t)(n0 + row) * K_TOT + kt + koff);
    asm volatile("global_load_async_to_lds_b128 %0, %1, off"
                 :: "v"(lds_addr), "v"(ga) : "memory");
  }
}

// ---------------------------------------------------------------------------
// Phase 2: GEMM
// ---------------------------------------------------------------------------
__global__ __launch_bounds__(256)
void sub4bit_gemm_kernel(const float* __restrict__ x,
                         const unsigned short* __restrict__ whi,
                         const unsigned short* __restrict__ wlo,
                         float* __restrict__ y) {
  __shared__ unsigned short sAhi[BM * APAD];
  __shared__ unsigned short sAlo[BM * APAD];
  __shared__ unsigned short sBhi[2][BN * APAD];
  __shared__ unsigned short sBlo[2][BN * APAD];

  const int t    = threadIdx.x;
  const int lane = t & 31;
  const int wave = t >> 5;
  const int wm   = wave & 3;   // 4 waves along M, 32 rows each
  const int wn   = wave >> 2;  // 2 waves along N, 64 cols each
  const int m0   = blockIdx.y * BM;
  const int n0   = blockIdx.x * BN;

  v8f acc[2][4] = {};

  // prologue: kick off B tile 0 (4 async instructions per wave)
  async_copy_tile(whi, &sBhi[0][0], n0, 0, t);
  async_copy_tile(wlo, &sBlo[0][0], n0, 0, t);

  for (int kt = 0, step = 0; kt < K_TOT; kt += BK, ++step) {
    const int buf = step & 1;

    // --- stage A tile: x[m0..+127][kt..+31] f32 -> hi/lo bf16 in LDS -------
#pragma unroll
    for (int j = 0; j < 4; ++j) {
      int f4  = j * 256 + t;        // 0..1023 float4 chunks (128 rows x 8)
      int row = f4 >> 3;
      int kk  = (f4 & 7) * 4;
      const float4 xv = *reinterpret_cast<const float4*>(
          x + (size_t)(m0 + row) * K_TOT + kt + kk);
      H4 h, l;
      split_bf16(xv.x, h.a, l.a);
      split_bf16(xv.y, h.b, l.b);
      split_bf16(xv.z, h.c, l.c);
      split_bf16(xv.w, h.d, l.d);
      unsigned base = row * APAD + kk;         // 8B aligned (80*row + 2*kk)
      *reinterpret_cast<H4*>(&sAhi[base]) = h;
      *reinterpret_cast<H4*>(&sAlo[base]) = l;
    }

    // --- double-buffered async B: issue next, then wait for current --------
    if (kt + BK < K_TOT) {
      async_copy_tile(whi, &sBhi[buf ^ 1][0], n0, kt + BK, t);
      async_copy_tile(wlo, &sBlo[buf ^ 1][0], n0, kt + BK, t);
      // 4 newest (next buf) may stay in flight; current buf's 4 are done.
      asm volatile("s_wait_asynccnt 0x4" ::: "memory");
      // prefetch next x tile into cache while we're at it
      const float* px = x + (size_t)(m0 + (t & 127)) * K_TOT + (kt + BK) + ((t >> 7) * 16);
      __builtin_prefetch(px, 0, 0);
    } else {
      asm volatile("s_wait_asynccnt 0x0" ::: "memory");
    }
    __syncthreads();

    // --- WMMA: 3-term bf16 split (hi*hi + hi*lo + lo*hi), f32 accumulate ---
    v16bf Ahi[2], Alo[2];
#pragma unroll
    for (int mf = 0; mf < 2; ++mf) {
      Ahi[mf] = load_frag_A(sAhi, 32 * wm + 16 * mf, lane);
      Alo[mf] = load_frag_A(sAlo, 32 * wm + 16 * mf, lane);
    }
#pragma unroll
    for (int nf = 0; nf < 4; ++nf) {
      v16bf Bhi = load_frag_B(&sBhi[buf][0], 64 * wn + 16 * nf, lane);
      v16bf Blo = load_frag_B(&sBlo[buf][0], 64 * wn + 16 * nf, lane);
#pragma unroll
      for (int mf = 0; mf < 2; ++mf) {
        acc[mf][nf] = wmma_bf16(Ahi[mf], Bhi, acc[mf][nf]);
        acc[mf][nf] = wmma_bf16(Ahi[mf], Blo, acc[mf][nf]);
        acc[mf][nf] = wmma_bf16(Alo[mf], Bhi, acc[mf][nf]);
      }
    }
    __syncthreads();
  }

  // --- epilogue: C layout = VGPR r: lanes 0-15 M=r, lanes 16-31 M=8+r ------
  const int mLaneOff = (lane < 16) ? 0 : 8;
  const int nLane    = lane & 15;
#pragma unroll
  for (int mf = 0; mf < 2; ++mf) {
#pragma unroll
    for (int nf = 0; nf < 4; ++nf) {
#pragma unroll
      for (int r = 0; r < 8; ++r) {
        int m = m0 + 32 * wm + 16 * mf + r + mLaneOff;
        int n = n0 + 64 * wn + 16 * nf + nLane;
        y[(size_t)m * N_TOT + n] = acc[mf][nf][r];
      }
    }
  }
}

extern "C" void kernel_launch(void* const* d_in, const int* in_sizes, int n_in,
                              void* d_out, int out_size, void* d_ws, size_t ws_size,
                              hipStream_t stream) {
  const float* x     = (const float*)d_in[0];
  const float* qg    = (const float*)d_in[1];
  const int*   codes = (const int*)d_in[2];
  float*       y     = (float*)d_out;

  // Workspace: Whi + Wlo bf16 tables, 2 * 4096*4096 * 2B = 64 MiB.
  unsigned short* whi = (unsigned short*)d_ws;
  unsigned short* wlo = whi + (size_t)N_TOT * K_TOT;

  dequant_split_kernel<<<dim3(N_TOT), 256, 0, stream>>>(qg, codes, whi, wlo);

  dim3 grid(N_TOT / BN, M_TOT / BM);   // 32 x 64 blocks
  sub4bit_gemm_kernel<<<grid, 256, 0, stream>>>(x, whi, wlo, y);
}
